// GGRUCell_88519275971179
// MI455X (gfx1250) — matrix operand order repaired and checked by reading, not compiled, not run on hardware
//
#include <hip/hip_runtime.h>
#include <hip/hip_bf16.h>

typedef __attribute__((ext_vector_type(16))) __bf16 v16bf;
typedef __attribute__((ext_vector_type(8)))  __bf16 v8bf;
typedef __attribute__((ext_vector_type(8)))  float  v8f;

#define BATCH 16384
#define HID   1024
#define KDIM  1024
#define LDSS  40                      // padded LDS row stride (bf16 units): 80B rows, 16B aligned
#define PLANE_ELEMS (128 * LDSS)      // one 128x32 tile plane (padded)
#define BUF_ELEMS   (4 * PLANE_ELEMS) // Ahi,Alo,Bhi,Blo
// double-buffered: 2 * 40960 B = 80 KB static LDS (WGP has 320 KB)

// -------- CDNA5 async global->LDS copy (ASYNCcnt path, inline asm per ISA §15.18.3) --------

__device__ __forceinline__ void async_b128(const __bf16* g, unsigned lds_off) {
  asm volatile("global_load_async_to_lds_b128 %0, %1, off"
               :: "v"(lds_off), "v"(g) : "memory");
}

__device__ __forceinline__ void wait_async_le8() {
  asm volatile("s_wait_asynccnt 0x8" ::: "memory");
}

__device__ __forceinline__ void wait_async_le0() {
  asm volatile("s_wait_asynccnt 0x0" ::: "memory");
}

// -------- fast transcendentals (v_exp_f32 + v_rcp_f32, no IEEE divide) --------

__device__ __forceinline__ float fast_sigmoid(float x) {
  return __builtin_amdgcn_rcpf(1.0f + __expf(-x));
}
__device__ __forceinline__ float fast_tanh(float x) {
  const float e = __expf(2.0f * x);            // +inf / 0 at extremes
  return 1.0f - 2.0f * __builtin_amdgcn_rcpf(e + 1.0f);  // -> +1 / -1, no NaN
}

// -------- fragment gather helpers --------

__device__ __forceinline__ v16bf frag2(const __bf16* p0, const __bf16* p1) {
  v8bf a = *(const v8bf*)p0;
  v8bf b = *(const v8bf*)p1;
  return __builtin_shufflevector(a, b, 0,1,2,3,4,5,6,7,8,9,10,11,12,13,14,15);
}

// One wave stages half a 128x32 plane (8 async b128 per k-step: 8 x 32 lanes x 16B = 4KB)
__device__ __forceinline__ void issue_plane(const __bf16* __restrict__ src,
                                            unsigned ldsPlaneBase, int kb, int qbase) {
  const int lane = threadIdx.x & 31;
#pragma unroll
  for (int i = 0; i < 8; ++i) {
    const int q   = qbase + i * 32 + lane;   // 16B-chunk index within plane half
    const int r   = q >> 2;                  // tile row 0..127
    const int col = (q & 3) * 8;             // bf16 col 0/8/16/24
    async_b128(src + (long)r * KDIM + kb + col,
               ldsPlaneBase + (unsigned)((r * LDSS + col) * 2));
  }
}

// Shared 128x128-tile mainloop. A: MxK row-major hi/lo planes; BT: NxK row-major
// (pre-transposed weights) hi/lo planes. 8 waves: wave pair (w>>1) stages plane w>>1;
// compute-wise wave (wm,wn) owns a 32x64 sub-tile = 2x4 16x16 accumulators.
__device__ __forceinline__ void gemm_mainloop(
    const __bf16* __restrict__ Ahi, const __bf16* __restrict__ Alo,
    const __bf16* __restrict__ BThi, const __bf16* __restrict__ BTlo,
    int rowBase, int colBase, __bf16* lds, v8f acc[2][4])
{
  const int tid   = threadIdx.x;
  const int lane  = tid & 31;
  const int wave  = tid >> 5;
  const int wm    = wave >> 1;   // 0..3
  const int wn    = wave & 1;    // 0..1
  const int laneL = lane & 15;
  const int khalf = lane >> 4;

  const unsigned ldsBase = (unsigned)(unsigned long long)(void*)lds;

  // staging assignment: wave pair -> plane, half selected by (wave&1)
  const int p = wave >> 1;       // 0:Ahi 1:Alo 2:Bhi 3:Blo
  const __bf16* planeSrc =
      (p == 0) ? Ahi  + (long)rowBase * KDIM :
      (p == 1) ? Alo  + (long)rowBase * KDIM :
      (p == 2) ? BThi + (long)colBase * KDIM :
                 BTlo + (long)colBase * KDIM;
  const unsigned planeOff = (unsigned)p * (PLANE_ELEMS * 2);
  const int qbase = (wave & 1) * 256;

  // prologue: prefetch k-tile 0 into buffer 0
  issue_plane(planeSrc, ldsBase + planeOff, 0, qbase);

  for (int kt = 0; kt < 32; ++kt) {
    const int cur = kt & 1;
    if (kt < 31) {
      // prefetch next k-tile into the other buffer (its readers finished at kt-1's barrier)
      issue_plane(planeSrc, ldsBase + (unsigned)(cur ^ 1) * (BUF_ELEMS * 2) + planeOff,
                  (kt + 1) * 32, qbase);
      wait_async_le8();          // in-order: current tile's 8 copies have landed
    } else {
      wait_async_le0();
    }
    __syncthreads();             // all waves' copies for `cur` are in LDS

    const __bf16* ldsAhi = lds + cur * BUF_ELEMS;
    const __bf16* ldsAlo = ldsAhi + PLANE_ELEMS;
    const __bf16* ldsBhi = ldsAhi + 2 * PLANE_ELEMS;
    const __bf16* ldsBlo = ldsAhi + 3 * PLANE_ELEMS;

    v16bf afh[2], afl[2], bfh[4], bfl[4];
#pragma unroll
    for (int mt = 0; mt < 2; ++mt) {
      // A frag: lane holds row M=laneL; K = {8*khalf..+7} U {8*khalf+16..+7}
      const __bf16* rh = ldsAhi + (wm*32 + mt*16 + laneL) * LDSS + khalf*8;
      const __bf16* rl = ldsAlo + (wm*32 + mt*16 + laneL) * LDSS + khalf*8;
      afh[mt] = frag2(rh, rh + 16);
      afl[mt] = frag2(rl, rl + 16);
    }
#pragma unroll
    for (int nt = 0; nt < 4; ++nt) {
      // B frag: lane holds column N=laneL; K = {8*khalf..+7} U {8*khalf+16..+7}
      const __bf16* rh = ldsBhi + (wn*64 + nt*16 + laneL) * LDSS + khalf*8;
      const __bf16* rl = ldsBlo + (wn*64 + nt*16 + laneL) * LDSS + khalf*8;
      bfh[nt] = frag2(rh, rh + 16);
      bfl[nt] = frag2(rl, rl + 16);
    }
#pragma unroll
    for (int mt = 0; mt < 2; ++mt) {
#pragma unroll
      for (int nt = 0; nt < 4; ++nt) {
        acc[mt][nt] = __builtin_amdgcn_wmma_f32_16x16x32_bf16(
            false, afh[mt], false, bfh[nt], (short)0, acc[mt][nt], false, false);
        acc[mt][nt] = __builtin_amdgcn_wmma_f32_16x16x32_bf16(
            false, afh[mt], false, bfl[nt], (short)0, acc[mt][nt], false, false);
        acc[mt][nt] = __builtin_amdgcn_wmma_f32_16x16x32_bf16(
            false, afl[mt], false, bfh[nt], (short)0, acc[mt][nt], false, false);
      }
    }
    __syncthreads();             // everyone done reading `cur` before it is overwritten
  }
}

// -------- prep kernels: f32 -> bf16 hi/lo split (and transposed variant) --------

__global__ __launch_bounds__(256) void split_rm(const float* __restrict__ src,
                                                __bf16* __restrict__ hi,
                                                __bf16* __restrict__ lo, long n) {
  long i = (long)blockIdx.x * blockDim.x + threadIdx.x;
  long stride = (long)gridDim.x * blockDim.x;
  for (; i < n; i += stride) {
    float  x = src[i];
    __bf16 h = (__bf16)x;
    hi[i] = h;
    lo[i] = (__bf16)(x - (float)h);
  }
}

// W is (K x N) row-major; write T (N x K) row-major. Writes coalesced along k.
__global__ __launch_bounds__(256) void transpose_split(const float* __restrict__ W,
                                                       __bf16* __restrict__ Thi,
                                                       __bf16* __restrict__ Tlo,
                                                       int K, int N) {
  int k = blockIdx.x * 256 + threadIdx.x;
  int n = blockIdx.y;
  if (k >= K) return;
  float  x = W[(long)k * N + n];
  __bf16 h = (__bf16)x;
  Thi[(long)n * K + k] = h;
  Tlo[(long)n * K + k] = (__bf16)(x - (float)h);
}

// -------- GEMM1: gates = sigmoid(prev@Wur + gate_inp); emit update (f32) and
//          a = state_inp*reset as bf16 hi/lo --------

__global__ __launch_bounds__(256) void gemm1_gates(
    const float* __restrict__ inp,
    const __bf16* __restrict__ Ahi, const __bf16* __restrict__ Alo,
    const __bf16* __restrict__ BThi, const __bf16* __restrict__ BTlo,
    float* __restrict__ update, __bf16* __restrict__ a_hi, __bf16* __restrict__ a_lo)
{
  __shared__ __bf16 lds[2 * BUF_ELEMS];
  const int rowBase = blockIdx.y * 128;
  const int colBase = blockIdx.x * 128;   // N = 2048

  v8f acc[2][4];
#pragma unroll
  for (int mt = 0; mt < 2; ++mt)
#pragma unroll
    for (int nt = 0; nt < 4; ++nt) { v8f z = {}; acc[mt][nt] = z; }

  gemm_mainloop(Ahi, Alo, BThi, BTlo, rowBase, colBase, lds, acc);

  const int lane = threadIdx.x & 31, wave = threadIdx.x >> 5;
  const int wm = wave >> 1, wn = wave & 1;
  const int laneL = lane & 15, khalf = lane >> 4;

#pragma unroll
  for (int mt = 0; mt < 2; ++mt) {
#pragma unroll
    for (int nt = 0; nt < 4; ++nt) {
      const int col = colBase + wn*64 + nt*16 + laneL;
#pragma unroll
      for (int r = 0; r < 8; ++r) {
        const int  row = rowBase + wm*32 + mt*16 + khalf*8 + r;
        const float x  = acc[mt][nt][r] + inp[(long)row * (3*HID) + HID + col];
        const float g  = fast_sigmoid(x);
        if (col < HID) {                       // update gate
          update[(long)row * HID + col] = g;
        } else {                               // reset gate -> a = state_inp * reset
          const int   nn = col - HID;
          const float s  = inp[(long)row * (3*HID) + nn];
          const float a  = s * g;
          const __bf16 h = (__bf16)a;
          a_hi[(long)row * HID + nn] = h;
          a_lo[(long)row * HID + nn] = (__bf16)(a - (float)h);
        }
      }
    }
  }
}

// -------- GEMM2: p = a@U; out = m*(tanh(p+state_inp)+update+(prev+(1-update))) + (1-m)*prev --------

__global__ __launch_bounds__(256) void gemm2_out(
    const float* __restrict__ inp, const float* __restrict__ prev,
    const float* __restrict__ mask,
    const __bf16* __restrict__ Ahi, const __bf16* __restrict__ Alo,
    const __bf16* __restrict__ BThi, const __bf16* __restrict__ BTlo,
    const float* __restrict__ update, float* __restrict__ out)
{
  __shared__ __bf16 lds[2 * BUF_ELEMS];
  const int rowBase = blockIdx.y * 128;
  const int colBase = blockIdx.x * 128;   // N = 1024

  v8f acc[2][4];
#pragma unroll
  for (int mt = 0; mt < 2; ++mt)
#pragma unroll
    for (int nt = 0; nt < 4; ++nt) { v8f z = {}; acc[mt][nt] = z; }

  gemm_mainloop(Ahi, Alo, BThi, BTlo, rowBase, colBase, lds, acc);

  const int lane = threadIdx.x & 31, wave = threadIdx.x >> 5;
  const int wm = wave >> 1, wn = wave & 1;
  const int laneL = lane & 15, khalf = lane >> 4;

#pragma unroll
  for (int mt = 0; mt < 2; ++mt) {
#pragma unroll
    for (int nt = 0; nt < 4; ++nt) {
      const int col = colBase + wn*64 + nt*16 + laneL;
#pragma unroll
      for (int r = 0; r < 8; ++r) {
        const int   row = rowBase + wm*32 + mt*16 + khalf*8 + r;
        const float s   = inp[(long)row * (3*HID) + col];        // state_inp
        const float t   = fast_tanh(acc[mt][nt][r] + s);
        const float u   = update[(long)row * HID + col];
        const float pv  = prev[(long)row * HID + col];
        const float nf  = t + u + (pv + (1.0f - u));
        const float m   = mask[row];
        out[(long)row * HID + col] = m * nf + (1.0f - m) * pv;
      }
    }
  }
}

// -------- launch --------

extern "C" void kernel_launch(void* const* d_in, const int* in_sizes, int n_in,
                              void* d_out, int out_size, void* d_ws, size_t ws_size,
                              hipStream_t stream) {
  const float* inp  = (const float*)d_in[0];   // (B, 3H)
  const float* prev = (const float*)d_in[1];   // (B, H)
  const float* mask = (const float*)d_in[2];   // (B,)
  const float* Wur  = (const float*)d_in[3];   // (H, 2H)
  const float* U    = (const float*)d_in[4];   // (H, H)
  float* out = (float*)d_out;

  char* ws = (char*)d_ws;
  size_t o = 0;
  __bf16* prev_hi = (__bf16*)(ws + o); o += (size_t)BATCH * HID * 2;
  __bf16* prev_lo = (__bf16*)(ws + o); o += (size_t)BATCH * HID * 2;
  __bf16* wurT_hi = (__bf16*)(ws + o); o += (size_t)HID * 2 * HID * 2;
  __bf16* wurT_lo = (__bf16*)(ws + o); o += (size_t)HID * 2 * HID * 2;
  __bf16* uT_hi   = (__bf16*)(ws + o); o += (size_t)HID * HID * 2;
  __bf16* uT_lo   = (__bf16*)(ws + o); o += (size_t)HID * HID * 2;
  __bf16* a_hi    = (__bf16*)(ws + o); o += (size_t)BATCH * HID * 2;
  __bf16* a_lo    = (__bf16*)(ws + o); o += (size_t)BATCH * HID * 2;
  float*  updw    = (float*)(ws + o);  o += (size_t)BATCH * HID * 4;

  // prep: hi/lo splits (weights transposed so B-operand tiles are K-contiguous)
  split_rm<<<2048, 256, 0, stream>>>(prev, prev_hi, prev_lo, (long)BATCH * HID);
  transpose_split<<<dim3(HID / 256, 2 * HID), 256, 0, stream>>>(Wur, wurT_hi, wurT_lo, HID, 2 * HID);
  transpose_split<<<dim3(HID / 256, HID),     256, 0, stream>>>(U,   uT_hi,   uT_lo,   HID, HID);

  // GEMM1: (B x H) @ (H x 2H), fused sigmoid epilogue
  gemm1_gates<<<dim3((2 * HID) / 128, BATCH / 128), 256, 0, stream>>>(
      inp, prev_hi, prev_lo, wurT_hi, wurT_lo, updw, a_hi, a_lo);

  // GEMM2: (B x H) @ (H x H), fused tanh + mask epilogue
  gemm2_out<<<dim3(HID / 128, BATCH / 128), 256, 0, stream>>>(
      inp, prev, mask, a_hi, a_lo, uT_hi, uT_lo, updw, out);
}
